// GroupedQueryAttention_75247827026231
// MI455X (gfx1250) — matrix operand order repaired and checked by reading, not compiled, not run on hardware
//
#include <hip/hip_runtime.h>
#include <hip/hip_bf16.h>

typedef __bf16 bf16;
typedef __attribute__((ext_vector_type(16))) __bf16 v16bf;
typedef __attribute__((ext_vector_type(8)))  float  v8f;
typedef __attribute__((ext_vector_type(4)))  unsigned int u32x4;

union FragU { v16bf v; u32x4 q[2]; };

__device__ inline v16bf load_frag(const bf16* p0, const bf16* p1) {
  FragU f;
  f.q[0] = *reinterpret_cast<const u32x4*>(p0);
  f.q[1] = *reinterpret_cast<const u32x4*>(p1);
  return f.v;
}

// CDNA5 async global->LDS copy: 16B per lane, tracked by ASYNCcnt.
__device__ inline void async_load16(unsigned lds_addr, const bf16* gsrc) {
  unsigned long long ga = (unsigned long long)gsrc;
  asm volatile("global_load_async_to_lds_b128 %0, %1, off"
               :: "v"(lds_addr), "v"(ga) : "memory");
}
__device__ inline void wait_async0() {
  asm volatile("s_wait_asynccnt 0" ::: "memory");
}

// ---------------------------------------------------------------------------
// Generic bf16 GEMM: C[M,N] = A[M,K] @ W[N,K]^T, fp32 accumulate.
// Block tile 128x64 (8 waves, 32x32 each), K staged in 64-chunks through
// double-buffered LDS via async global->LDS loads.
// ---------------------------------------------------------------------------
#define LDA 72

__global__ __launch_bounds__(256) void gemm_bf16_nt(
    const bf16* __restrict__ A, const bf16* __restrict__ W,
    float* __restrict__ C, int M, int N, int K)
{
  __shared__ __align__(16) bf16 As[2][128 * LDA];
  __shared__ __align__(16) bf16 Bs[2][64 * LDA];
  const int tid  = threadIdx.x;
  const int lane = tid & 31;
  const int wv   = tid >> 5;          // 0..7
  const int half = lane >> 4;         // 0/1
  const int l16  = lane & 15;
  const int m0 = blockIdx.y * 128;
  const int n0 = blockIdx.x * 64;
  const int mw = (wv & 3) * 32;       // wave row offset inside tile
  const int nw = (wv >> 2) * 32;      // wave col offset inside tile

  v8f c00 = {}, c01 = {}, c10 = {}, c11 = {};

  auto stage = [&](int buf, int k0) {
    #pragma unroll
    for (int i = 0; i < 4; ++i) {     // A: 128 rows x 8 chunks = 1024
      int idx = tid + i * 256;
      int r = idx >> 3, cc = idx & 7;
      async_load16((unsigned)(size_t)&As[buf][r * LDA + cc * 8],
                   &A[(size_t)(m0 + r) * K + k0 + cc * 8]);
    }
    #pragma unroll
    for (int i = 0; i < 2; ++i) {     // B: 64 rows x 8 chunks = 512
      int idx = tid + i * 256;
      int r = idx >> 3, cc = idx & 7;
      async_load16((unsigned)(size_t)&Bs[buf][r * LDA + cc * 8],
                   &W[(size_t)(n0 + r) * K + k0 + cc * 8]);
    }
  };

  stage(0, 0);
  wait_async0();
  __syncthreads();

  int cur = 0;
  for (int k0 = 0; k0 < K; k0 += 64) {
    int nxt = cur ^ 1;
    if (k0 + 64 < K) stage(nxt, k0 + 64);   // overlap copy with compute
    #pragma unroll
    for (int kk = 0; kk < 64; kk += 32) {
      const bf16* a0p = &As[cur][(mw + l16) * LDA + kk];
      const bf16* a1p = &As[cur][(mw + 16 + l16) * LDA + kk];
      v16bf a0 = load_frag(a0p + half * 8, a0p + 16 + half * 8);
      v16bf a1 = load_frag(a1p + half * 8, a1p + 16 + half * 8);
      const bf16* b0p = &Bs[cur][(nw + l16) * LDA + kk];
      const bf16* b1p = &Bs[cur][(nw + 16 + l16) * LDA + kk];
      v16bf b0 = load_frag(b0p + half * 8, b0p + 16 + half * 8);
      v16bf b1 = load_frag(b1p + half * 8, b1p + 16 + half * 8);
      c00 = __builtin_amdgcn_wmma_f32_16x16x32_bf16(false, a0, false, b0, (short)0, c00, false, false);
      c01 = __builtin_amdgcn_wmma_f32_16x16x32_bf16(false, a0, false, b1, (short)0, c01, false, false);
      c10 = __builtin_amdgcn_wmma_f32_16x16x32_bf16(false, a1, false, b0, (short)0, c10, false, false);
      c11 = __builtin_amdgcn_wmma_f32_16x16x32_bf16(false, a1, false, b1, (short)0, c11, false, false);
    }
    wait_async0();
    __syncthreads();
    cur = nxt;
  }

  const int row0 = m0 + mw + half * 8;
  const int col0 = n0 + nw + l16;
  #pragma unroll
  for (int r = 0; r < 8; ++r) {
    C[(size_t)(row0 + r) * N + col0]           = c00[r];
    C[(size_t)(row0 + r) * N + col0 + 16]      = c01[r];
    C[(size_t)(row0 + 16 + r) * N + col0]      = c10[r];
    C[(size_t)(row0 + 16 + r) * N + col0 + 16] = c11[r];
  }
}

// ---------------------------------------------------------------------------
// Elementwise conversion / RoPE / V transpose
// ---------------------------------------------------------------------------
__global__ void cvt_bf16_kernel(const float* __restrict__ s, bf16* __restrict__ d, int n) {
  int i = blockIdx.x * blockDim.x + threadIdx.x;
  if (i < n) d[i] = (bf16)s[i];
}

__global__ void rope_q_kernel(const float* __restrict__ q, const float* __restrict__ fc,
                              bf16* __restrict__ out) {
  int idx = blockIdx.x * blockDim.x + threadIdx.x;   // (s*32 + h)*64 + d2
  if (idx >= 2048 * 32 * 64) return;
  int d2 = idx & 63;
  int sh = idx >> 6;
  int h = sh & 31;
  int s = sh >> 5;
  size_t base = (size_t)s * 4096 + h * 128 + d2 * 2;
  float a = q[base], b = q[base + 1];
  float cs = fc[(s * 64 + d2) * 2], sn = fc[(s * 64 + d2) * 2 + 1];
  out[base]     = (bf16)(a * cs - b * sn);
  out[base + 1] = (bf16)(a * sn + b * cs);
}

__global__ void rope_k_kernel(const float* __restrict__ k, const float* __restrict__ fc,
                              bf16* __restrict__ out) {   // out: (KV, S, 128)
  int idx = blockIdx.x * blockDim.x + threadIdx.x;   // (s*8 + g)*64 + d2
  if (idx >= 2048 * 8 * 64) return;
  int d2 = idx & 63;
  int sg = idx >> 6;
  int g = sg & 7;
  int s = sg >> 3;
  size_t src = (size_t)s * 1024 + g * 128 + d2 * 2;
  float a = k[src], b = k[src + 1];
  float cs = fc[(s * 64 + d2) * 2], sn = fc[(s * 64 + d2) * 2 + 1];
  size_t dst = ((size_t)g * 2048 + s) * 128 + d2 * 2;
  out[dst]     = (bf16)(a * cs - b * sn);
  out[dst + 1] = (bf16)(a * sn + b * cs);
}

__global__ void vtrans_kernel(const float* __restrict__ v, bf16* __restrict__ vt) {
  int idx = blockIdx.x * blockDim.x + threadIdx.x;   // s*1024 + (g*128+d)
  if (idx >= 2048 * 1024) return;
  int c = idx & 1023;
  int s = idx >> 10;
  int g = c >> 7, d = c & 127;
  vt[((size_t)g * 128 + d) * 2048 + s] = (bf16)v[idx];
}

// ---------------------------------------------------------------------------
// Flash attention: block = (head, 128 query rows), 8 waves x 16 rows.
// Key blocks of 32 staged in LDS via async loads. Online softmax, fp32 accum.
// ---------------------------------------------------------------------------
#define KSTR 136
#define VSTR 40
#define PSTR 40

__global__ __launch_bounds__(256) void attn_kernel(
    const bf16* __restrict__ Q,   // (S, 4096) row-major, head-contiguous
    const bf16* __restrict__ Kc,  // (KV, S, 128)
    const bf16* __restrict__ Vt,  // (KV, 128, S)
    bf16* __restrict__ O)         // (S, 4096)
{
  __shared__ __align__(16) bf16 Ks[32 * KSTR];
  __shared__ __align__(16) bf16 Vs[128 * VSTR];
  __shared__ __align__(16) bf16 Ps[8 * 16 * PSTR];

  const int tid  = threadIdx.x;
  const int lane = tid & 31;
  const int wv   = tid >> 5;
  const int half = lane >> 4;
  const int l16  = lane & 15;
  const int h = blockIdx.y;
  const int g = h >> 2;
  const int qr = blockIdx.x * 128 + wv * 16;   // first query row of this wave

  // Q fragments: 16 rows x 128, 4 chunks of K=32
  v16bf qf[4];
  #pragma unroll
  for (int c = 0; c < 4; ++c) {
    const bf16* p = &Q[(size_t)(qr + l16) * 4096 + h * 128 + c * 32];
    qf[c] = load_frag(p + half * 8, p + 16 + half * 8);
  }

  v8f zero = {};
  v8f o[8];
  #pragma unroll
  for (int i = 0; i < 8; ++i) o[i] = zero;
  float mrow[8], lrow[8];
  #pragma unroll
  for (int r = 0; r < 8; ++r) { mrow[r] = -3.0e38f; lrow[r] = 0.0f; }

  const int nkb = blockIdx.x * 4 + 4;          // key blocks of 32 (causal bound)
  for (int kb = 0; kb < nkb; ++kb) {
    const int k0 = kb * 32;
    // stage K block, first 64 of 128 dims (32 x 64): 256 chunks of 16B
    {
      int r = tid >> 3, cc = tid & 7;
      async_load16((unsigned)(size_t)&Ks[r * KSTR + cc * 8],
                   &Kc[((size_t)g * 2048 + k0 + r) * 128 + cc * 8]);
    }
    // stage Vt block (128 x 32): 512 chunks of 16B
    #pragma unroll
    for (int i = 0; i < 2; ++i) {
      int idx = tid + i * 256;
      int rv = idx >> 2, cv = idx & 3;
      async_load16((unsigned)(size_t)&Vs[rv * VSTR + cv * 8],
                   &Vt[((size_t)g * 128 + rv) * 2048 + k0 + cv * 8]);
    }
    wait_async0();
    __syncthreads();

    if (k0 <= qr + 15) {
      // scores: two 16-key tiles, K = HD = 128 in 4 WMMA each
      v8f s[2];
      #pragma unroll
      for (int t = 0; t < 2; ++t) {
        v8f acc = zero;
        #pragma unroll
        for (int c = 0; c < 4; ++c) {
          const bf16* kp = &Ks[(t * 16 + l16) * KSTR + c * 32];
          v16bf b = load_frag(kp + half * 8, kp + 16 + half * 8);
          acc = __builtin_amdgcn_wmma_f32_16x16x32_bf16(false, qf[c], false, b, (short)0, acc, false, false);
        }
        s[t] = acc;
      }
      // scale + causal mask
      const float SC = 0.08838834764831845f;   // 1/sqrt(128)
      #pragma unroll
      for (int t = 0; t < 2; ++t) {
        int key = k0 + t * 16 + l16;
        #pragma unroll
        for (int r = 0; r < 8; ++r) {
          int rowg = qr + half * 8 + r;
          float v = s[t][r] * SC;
          s[t][r] = (key > rowg) ? -1.0e30f : v;
        }
      }
      // online softmax (row stats live across the 16 lanes of each half)
      float fac[8];
      #pragma unroll
      for (int r = 0; r < 8; ++r) {
        float mx = fmaxf(s[0][r], s[1][r]);
        #pragma unroll
        for (int off = 1; off < 16; off <<= 1)
          mx = fmaxf(mx, __shfl_xor(mx, off, 32));
        float mnew = fmaxf(mrow[r], mx);
        fac[r] = __expf(mrow[r] - mnew);
        float p0 = __expf(s[0][r] - mnew);
        float p1 = __expf(s[1][r] - mnew);
        s[0][r] = p0; s[1][r] = p1;
        float ps = p0 + p1;
        #pragma unroll
        for (int off = 1; off < 16; off <<= 1)
          ps += __shfl_xor(ps, off, 32);
        lrow[r] = lrow[r] * fac[r] + ps;
        mrow[r] = mnew;
      }
      // rescale accumulators
      #pragma unroll
      for (int dt = 0; dt < 8; ++dt)
        #pragma unroll
        for (int r = 0; r < 8; ++r)
          o[dt][r] *= fac[r];
      // probs -> per-wave LDS (row-major 16 x 32), then rebuild A fragment
      bf16* pw = &Ps[wv * 16 * PSTR];
      #pragma unroll
      for (int t = 0; t < 2; ++t)
        #pragma unroll
        for (int r = 0; r < 8; ++r)
          pw[(half * 8 + r) * PSTR + t * 16 + l16] = (bf16)s[t][r];
      const bf16* pr = &Ps[wv * 16 * PSTR + l16 * PSTR];
      v16bf pa = load_frag(pr + half * 8, pr + 16 + half * 8);
      // PV: 8 column tiles of 16 over HD=128
      #pragma unroll
      for (int dt = 0; dt < 8; ++dt) {
        const bf16* vp = &Vs[(dt * 16 + l16) * VSTR];
        v16bf b = load_frag(vp + half * 8, vp + 16 + half * 8);
        o[dt] = __builtin_amdgcn_wmma_f32_16x16x32_bf16(false, pa, false, b, (short)0, o[dt], false, false);
      }
    }
    __syncthreads();
  }

  #pragma unroll
  for (int r = 0; r < 8; ++r) lrow[r] = 1.0f / lrow[r];
  #pragma unroll
  for (int dt = 0; dt < 8; ++dt) {
    #pragma unroll
    for (int r = 0; r < 8; ++r) {
      size_t idx = (size_t)(qr + half * 8 + r) * 4096 + h * 128 + dt * 16 + l16;
      O[idx] = (bf16)(o[dt][r] * lrow[r]);
    }
  }
}

// ---------------------------------------------------------------------------
extern "C" void kernel_launch(void* const* d_in, const int* in_sizes, int n_in,
                              void* d_out, int out_size, void* d_ws, size_t ws_size,
                              hipStream_t stream) {
  (void)in_sizes; (void)n_in; (void)out_size; (void)ws_size;
  const float* x  = (const float*)d_in[0];
  const float* fc = (const float*)d_in[1];
  // d_in[2] = mask (pure causal -1e9; reproduced analytically)
  const float* wq = (const float*)d_in[3];
  const float* wk = (const float*)d_in[4];
  const float* wv = (const float*)d_in[5];
  const float* wo = (const float*)d_in[6];
  float* out = (float*)d_out;

  char* ws = (char*)d_ws;
  size_t off = 0;
  auto alloc = [&](size_t bytes) -> void* {
    void* p = ws + off;
    off += (bytes + 255) & ~(size_t)255;
    return p;
  };
  bf16*  xb  = (bf16*) alloc(2048ull * 4096 * 2);
  bf16*  wqb = (bf16*) alloc(4096ull * 4096 * 2);
  bf16*  wkb = (bf16*) alloc(1024ull * 4096 * 2);
  bf16*  wvb = (bf16*) alloc(1024ull * 4096 * 2);
  bf16*  wob = (bf16*) alloc(4096ull * 4096 * 2);
  float* qf  = (float*)alloc(2048ull * 4096 * 4);
  float* kf  = (float*)alloc(2048ull * 1024 * 4);
  float* vf  = (float*)alloc(2048ull * 1024 * 4);
  bf16*  qb2 = (bf16*) alloc(2048ull * 4096 * 2);
  bf16*  kb2 = (bf16*) alloc(2048ull * 1024 * 2);
  bf16*  vtt = (bf16*) alloc(2048ull * 1024 * 2);
  bf16*  ob  = (bf16*) alloc(2048ull * 4096 * 2);

  const int TPB = 256;
  int n;
  n = 2048 * 4096; cvt_bf16_kernel<<<(n + TPB - 1) / TPB, TPB, 0, stream>>>(x,  xb,  n);
  n = 4096 * 4096; cvt_bf16_kernel<<<(n + TPB - 1) / TPB, TPB, 0, stream>>>(wq, wqb, n);
  n = 1024 * 4096; cvt_bf16_kernel<<<(n + TPB - 1) / TPB, TPB, 0, stream>>>(wk, wkb, n);
  n = 1024 * 4096; cvt_bf16_kernel<<<(n + TPB - 1) / TPB, TPB, 0, stream>>>(wv, wvb, n);
  n = 4096 * 4096; cvt_bf16_kernel<<<(n + TPB - 1) / TPB, TPB, 0, stream>>>(wo, wob, n);

  gemm_bf16_nt<<<dim3(4096 / 64, 2048 / 128), TPB, 0, stream>>>(xb, wqb, qf, 2048, 4096, 4096);
  gemm_bf16_nt<<<dim3(1024 / 64, 2048 / 128), TPB, 0, stream>>>(xb, wkb, kf, 2048, 1024, 4096);
  gemm_bf16_nt<<<dim3(1024 / 64, 2048 / 128), TPB, 0, stream>>>(xb, wvb, vf, 2048, 1024, 4096);

  n = 2048 * 32 * 64; rope_q_kernel<<<(n + TPB - 1) / TPB, TPB, 0, stream>>>(qf, fc, qb2);
  n = 2048 * 8 * 64;  rope_k_kernel<<<(n + TPB - 1) / TPB, TPB, 0, stream>>>(kf, fc, kb2);
  n = 2048 * 1024;    vtrans_kernel<<<(n + TPB - 1) / TPB, TPB, 0, stream>>>(vf, vtt);

  attn_kernel<<<dim3(16, 32), TPB, 0, stream>>>(qb2, kb2, vtt, ob);

  gemm_bf16_nt<<<dim3(4096 / 64, 2048 / 128), TPB, 0, stream>>>(ob, wob, out, 2048, 4096, 4096);
}
